// conv1d_resnet_block_knnGraph_11733850653060
// MI455X (gfx1250) — matrix operand order repaired and checked by reading, not compiled, not run on hardware
//
#include <hip/hip_runtime.h>

typedef __attribute__((ext_vector_type(16))) __bf16 v16bf;
typedef __attribute__((ext_vector_type(8)))  __bf16 v8bf;
typedef __attribute__((ext_vector_type(8)))  float  v8f;
typedef __attribute__((ext_vector_type(2)))  float  v2f;

#define BATCH 8
#define CH    128
#define NPT   2048
#define KNNK  10
#define CH2   256

__device__ __forceinline__ void wave_fence() {
  // intra-wave LDS exchange: LDS ops from one wave complete in order (ISA §7.3);
  // we only need to stop the compiler from reordering around the exchange.
  asm volatile("" ::: "memory");
  __builtin_amdgcn_wave_barrier();
  asm volatile("" ::: "memory");
}

// ---------- fragment loaders (layouts per CDNA5 ISA §7.12.2) ----------
// 16-bit A 16x32: lane(0-15)=row M, half=lane>>4; elems 0..7 -> K=8*half+j, 8..15 -> K=16+8*half+(j-8)
__device__ inline v16bf load_a_bf16(const __bf16* __restrict__ row, int c0, int half) {
  v8bf lo = *(const v8bf*)(row + c0 + 8 * half);
  v8bf hi = *(const v8bf*)(row + c0 + 16 + 8 * half);
  v16bf r;
#pragma unroll
  for (int t = 0; t < 8; ++t) { r[t] = lo[t]; r[8 + t] = hi[t]; }
  return r;
}
// 16-bit B 32x16: lane(0-15)=col N; elems 0..7 -> K=8*half+j, 8..15 -> K=16+8*half+(j-8)
__device__ inline v16bf load_b_bf16(const __bf16* __restrict__ row, int c0, int half) {
  v8bf lo = *(const v8bf*)(row + c0 + 8 * half);
  v8bf hi = *(const v8bf*)(row + c0 + 16 + 8 * half);
  v16bf r;
#pragma unroll
  for (int t = 0; t < 8; ++t) { r[t] = lo[t]; r[8 + t] = hi[t]; }
  return r;
}

// ---------- W concat prep: wcat = [W[:, :C] | W[:, C:] - W[:, :C]] ----------
__global__ __launch_bounds__(256) void wcat_kernel(const float* __restrict__ W1,
                                                   const float* __restrict__ W2,
                                                   float* __restrict__ wcat) {
  const int layer = blockIdx.y;
  const float* W = layer ? W2 : W1;
  int i = blockIdx.x * 256 + threadIdx.x;         // over CH*CH2 = 32768
  int o = i >> 8, c = i & 255;
  float wa = W[o * CH2 + (c & (CH - 1))];
  float v = (c < CH) ? wa : (W[o * CH2 + c] - wa);
  wcat[(size_t)layer * CH * CH2 + i] = v;
}

// ---------- transpose h[b][c][n] -> hTf/hTb [b][n][c] (f32 + bf16) ----------
__global__ __launch_bounds__(256) void transpose_kernel(const float* __restrict__ h,
                                                        float* __restrict__ hTf,
                                                        __bf16* __restrict__ hTb) {
  __shared__ float t[16][17];
  const int b = blockIdx.z, c0 = blockIdx.y * 16, n0 = blockIdx.x * 16;
  const int tx = threadIdx.x, ty = threadIdx.y;
  t[ty][tx] = h[((size_t)b * CH + c0 + ty) * NPT + n0 + tx];
  __syncthreads();
  float v = t[tx][ty];
  size_t oidx = ((size_t)b * NPT + n0 + ty) * CH + c0 + tx;
  hTf[oidx] = v;
  hTb[oidx] = (__bf16)v;
}

// ---------- xx[b][n] = sum_c h^2 ----------
__global__ __launch_bounds__(256) void rowsq_kernel(const float* __restrict__ hTf,
                                                    float* __restrict__ xx) {
  const int wave = threadIdx.x >> 5, lane = threadIdx.x & 31;
  const int b = blockIdx.y, n = blockIdx.x * 8 + wave;
  const float* row = hTf + ((size_t)b * NPT + n) * CH;
  float s = 0.f;
#pragma unroll
  for (int t = 0; t < 4; ++t) { float v = row[lane + 32 * t]; s += v * v; }
#pragma unroll
  for (int off = 16; off > 0; off >>= 1) s += __shfl_xor(s, off, 32);
  if (lane == 0) xx[b * NPT + n] = s;
}

// ---------- fused pairwise-distance GEMM (bf16 WMMA) + top-10 selection ----------
// Wave owns a 16-row strip. Lane l and lane l+16 co-own row (l&15): lane half 0
// scans tile cols 0..7, half 1 scans cols 8..15, each keeping a private sorted
// top-10; the two half-lists are merged once at the end through LDS.
__global__ __launch_bounds__(128) void knn_kernel(const __bf16* __restrict__ hTb,
                                                  const float* __restrict__ xx,
                                                  int* __restrict__ knn) {
  const int wave = threadIdx.x >> 5, lane = threadIdx.x & 31;
  const int b = blockIdx.y;
  const int n0 = blockIdx.x * 64 + wave * 16;
  const int ml = lane & 15, half = lane >> 4;
  const int jbase = half * 8;

  const __bf16* hb = hTb + (size_t)b * NPT * CH;
  const __bf16* arow = hb + (size_t)(n0 + ml) * CH;

  v16bf afrag[4];
#pragma unroll
  for (int kk = 0; kk < 4; ++kk) afrag[kk] = load_a_bf16(arow, kk * 32, half);

  float topv[KNNK]; int topi[KNNK];
#pragma unroll
  for (int t = 0; t < KNNK; ++t) { topv[t] = -3.0e38f; topi[t] = 0; }

  __shared__ float tile[4][16][17];             // padded row pitch: conflict-free strided reads
  __shared__ float mv[4][2][16][KNNK];
  __shared__ int   mi[4][2][16][KNNK];

  for (int m0 = 0; m0 < NPT; m0 += 16) {
    const __bf16* brow = hb + (size_t)(m0 + ml) * CH;
    if (m0 + 16 < NPT)                          // cover load latency of next tile
      __builtin_prefetch(hb + (size_t)(m0 + 16 + ml) * CH, 0, 0);
    v8f acc = {};
#pragma unroll
    for (int kk = 0; kk < 4; ++kk) {
      v16bf bfrag = load_b_bf16(brow, kk * 32, half);
      acc = __builtin_amdgcn_wmma_f32_16x16x32_bf16(false, afrag[kk], false, bfrag,
                                                    (short)0, acc, false, false);
    }
    // ranking score: 2*inner - xx[m]   (-xx[n] is row-constant, irrelevant to top-k)
    const float xm = xx[b * NPT + m0 + ml];
#pragma unroll
    for (int r = 0; r < 8; ++r) tile[wave][r + 8 * half][ml] = 2.0f * acc[r] - xm;
    wave_fence();
#pragma unroll
    for (int j = 0; j < 8; ++j) {
      float val = tile[wave][ml][jbase + j];
      int cand = m0 + jbase + j;
#pragma unroll
      for (int p = 0; p < KNNK; ++p) {          // branch-free insertion network (registers only)
        bool gt = val > topv[p];
        float ov = topv[p]; int oi = topi[p];
        topv[p] = gt ? val : ov;  topi[p] = gt ? cand : oi;
        val     = gt ? ov  : val; cand    = gt ? oi   : cand;
      }
    }
    wave_fence();
  }

  // merge the two sorted half-lists for each row
#pragma unroll
  for (int t = 0; t < KNNK; ++t) { mv[wave][half][ml][t] = topv[t]; mi[wave][half][ml][t] = topi[t]; }
  wave_fence();
  if (lane < 16) {
    int* op = knn + ((size_t)b * NPT + n0 + ml) * KNNK;
    int ia = 0, ib = 0;
#pragma unroll
    for (int t = 0; t < KNNK; ++t) {
      float va = mv[wave][0][ml][ia], vb = mv[wave][1][ml][ib];
      bool takeB = vb > va;                     // ties -> lower column group (lower index)
      op[t] = takeB ? mi[wave][1][ml][ib] : mi[wave][0][ml][ia];
      ia += takeB ? 0 : 1;
      ib += takeB ? 1 : 0;
    }
  }
}

// ---------- gather: ZT[b][n][0:C]=mean_k nbr, ZT[b][n][C:2C]=ctr ----------
__global__ __launch_bounds__(128) void gather_kernel(const float* __restrict__ hTf,
                                                     const int* __restrict__ knn,
                                                     float* __restrict__ ZT) {
  const int b = blockIdx.y, n = blockIdx.x, c = threadIdx.x;
  const int* ip = knn + ((size_t)b * NPT + n) * KNNK;
  const float* hb = hTf + (size_t)b * NPT * CH;
  float s = 0.f;
#pragma unroll
  for (int k = 0; k < KNNK; ++k) s += hb[(size_t)ip[k] * CH + c];
  float* zp = ZT + ((size_t)b * NPT + n) * CH2;
  zp[c] = s * (1.0f / KNNK);
  zp[CH + c] = hb[(size_t)n * CH + c];
}

// ---------- EdgeConv GEMM: ypre[b][o][n] = wcat(128x256) @ ZT^T + bias  (f32 WMMA) ----------
__global__ __launch_bounds__(256) void edge_gemm_kernel(const float* __restrict__ wcat,
                                                        const float* __restrict__ ZT,
                                                        const float* __restrict__ bias,
                                                        float* __restrict__ ypre) {
  const int wave = threadIdx.x >> 5, lane = threadIdx.x & 31;
  const int b = blockIdx.y, n0 = blockIdx.x * 16, o0 = wave * 16;
  const int ml = lane & 15, half = lane >> 4;

  const float* arow = wcat + (size_t)(o0 + ml) * CH2;               // A: 16x4 f32, elem j -> K=2*half+j
  const float* brow = ZT + ((size_t)b * NPT + n0 + ml) * CH2;       // B: 4x16 f32, same K striping

  v8f acc = {};
#pragma unroll 16
  for (int kk = 0; kk < 64; ++kk) {
    int c = kk * 4 + 2 * half;
    v2f a = *(const v2f*)(arow + c);
    v2f bb = *(const v2f*)(brow + c);
    acc = __builtin_amdgcn_wmma_f32_16x16x4_f32(false, a, false, bb, (short)0, acc,
                                                false, false);
  }
#pragma unroll
  for (int r = 0; r < 8; ++r) {
    int o = o0 + r + 8 * half;
    ypre[((size_t)b * CH + o) * NPT + n0 + ml] = acc[r] + bias[o];
  }
}

// ---------- per-(b,o) mean/var(ddof=1) over N, ReLU, optional residual ----------
__global__ __launch_bounds__(256) void norm_relu_kernel(const float* __restrict__ y,
                                                        const float* __restrict__ resid,
                                                        float* __restrict__ outp) {
  const int b = blockIdx.y, o = blockIdx.x;
  const float* row = y + ((size_t)b * CH + o) * NPT;
  float s = 0.f, ss = 0.f;
  for (int i = threadIdx.x; i < NPT; i += 256) { float v = row[i]; s += v; ss += v * v; }
  __shared__ float sbuf[256], qbuf[256];
  sbuf[threadIdx.x] = s; qbuf[threadIdx.x] = ss;
  __syncthreads();
  for (int off = 128; off > 0; off >>= 1) {
    if (threadIdx.x < off) {
      sbuf[threadIdx.x] += sbuf[threadIdx.x + off];
      qbuf[threadIdx.x] += qbuf[threadIdx.x + off];
    }
    __syncthreads();
  }
  float mean = sbuf[0] * (1.0f / NPT);
  float var = (qbuf[0] - (float)NPT * mean * mean) * (1.0f / (NPT - 1));
  float inv = rsqrtf(var + 1e-3f);
  const float* rrow = resid ? resid + ((size_t)b * CH + o) * NPT : nullptr;
  float* orow = outp + ((size_t)b * CH + o) * NPT;
  for (int i = threadIdx.x; i < NPT; i += 256) {
    float v = (row[i] - mean) * inv;
    v = v > 0.f ? v : 0.f;
    if (rrow) v += rrow[i];
    orow[i] = v;
  }
}

extern "C" void kernel_launch(void* const* d_in, const int* in_sizes, int n_in,
                              void* d_out, int out_size, void* d_ws, size_t ws_size,
                              hipStream_t stream) {
  const float* x  = (const float*)d_in[0];
  const float* W1 = (const float*)d_in[1];
  const float* b1 = (const float*)d_in[2];
  const float* W2 = (const float*)d_in[3];
  const float* b2 = (const float*)d_in[4];

  char* p = (char*)d_ws;
  float*  hTf  = (float*)p;  p += (size_t)BATCH * NPT * CH * 4;        // 8 MB
  __bf16* hTb  = (__bf16*)p; p += (size_t)BATCH * NPT * CH * 2;        // 4 MB
  float*  xx   = (float*)p;  p += (size_t)BATCH * NPT * 4;             // 64 KB
  int*    knn  = (int*)p;    p += (size_t)BATCH * NPT * KNNK * 4;      // 640 KB
  float*  ZT   = (float*)p;  p += (size_t)BATCH * NPT * CH2 * 4;       // 16 MB
  float*  ypre = (float*)p;  p += (size_t)BATCH * CH * NPT * 4;        // 8 MB
  float*  h2   = (float*)p;  p += (size_t)BATCH * CH * NPT * 4;        // 8 MB
  float*  wcat = (float*)p;  p += (size_t)2 * CH * CH2 * 4;            // 256 KB
  (void)ws_size; (void)in_sizes; (void)n_in; (void)out_size;

  wcat_kernel<<<dim3(128, 2), 256, 0, stream>>>(W1, W2, wcat);

  for (int layer = 0; layer < 2; ++layer) {
    const float* h    = layer ? h2 : x;
    const float* bias = layer ? b2 : b1;
    const float* wl   = wcat + (size_t)layer * CH * CH2;
    float* dst        = layer ? (float*)d_out : h2;
    const float* res  = layer ? x : nullptr;

    transpose_kernel<<<dim3(NPT / 16, CH / 16, BATCH), dim3(16, 16), 0, stream>>>(h, hTf, hTb);
    rowsq_kernel<<<dim3(NPT / 8, BATCH), 256, 0, stream>>>(hTf, xx);
    knn_kernel<<<dim3(NPT / 64, BATCH), 128, 0, stream>>>(hTb, xx, knn);
    gather_kernel<<<dim3(NPT, BATCH), 128, 0, stream>>>(hTf, knn, ZT);
    edge_gemm_kernel<<<dim3(NPT / 16, BATCH), 256, 0, stream>>>(wl, ZT, bias, ypre);
    norm_relu_kernel<<<dim3(CH, BATCH), 256, 0, stream>>>(ypre, res, dst);
  }
}
